// S2ViT_56977036149089
// MI455X (gfx1250) — matrix operand order repaired and checked by reading, not compiled, not run on hardware
//
#include <hip/hip_runtime.h>
#include <math.h>

// ---------------------------------------------------------------------------
// CDNA5 (gfx1250) wave32 WMMA types & fragment helpers.
// Layouts (A/B: contiguous k order chosen by this code, not the ISA table order; A and B match):
//   A (f16 16x32): lane m=l&15; lanes<16 hold K 0..15,
//                  lanes>=16 hold K 16..31.
//   B (f16 32x16): lane n=l&15; lanes<16 hold K 0..15, lanes>=16 K 16..31.
//   C/D (f32 16x16): lane n=l&15; VGPR v holds row m = v + 8*(l>>4).
// All B operands are stored transposed (B^T, [N x K] row-major) so every
// fragment load is two contiguous 16-byte loads per lane - no per-element
// bounds checks, no EXEC-masked scalar loads.
// ---------------------------------------------------------------------------
typedef __attribute__((ext_vector_type(16))) _Float16 v16h;
typedef __attribute__((ext_vector_type(8)))  _Float16 v8h;
typedef __attribute__((ext_vector_type(8)))  float    v8f;

__device__ __forceinline__ v8f wmma16(v16h a, v16h b, v8f c) {
  return __builtin_amdgcn_wmma_f32_16x16x32_f16(false, a, false, b,
                                                (short)0, c, false, false);
}

// A: 16 x K row-major (halves). Requires 16B-aligned rows (lda % 8 == 0).
__device__ __forceinline__ v16h load_a_frag(const _Float16* A, int lda, int k0) {
  const int lane = threadIdx.x & 31;
  const _Float16* r = A + (lane & 15) * lda + k0 + ((lane >> 4) << 4);
  v8h lo = *(const v8h*)r;
  v8h hi = *(const v8h*)(r + 8);
  return __builtin_shufflevector(lo, hi, 0, 1, 2, 3, 4, 5, 6, 7,
                                 8, 9, 10, 11, 12, 13, 14, 15);
}

// B from transposed storage: BT is [N x K] row-major, B[k][n] = BT[n*ldk+k].
// Requires ldk % 16 == 0 (rows 32B aligned).
__device__ __forceinline__ v16h load_bt_frag(const _Float16* BT, int ldk,
                                             int k0, int n0) {
  const int lane = threadIdx.x & 31;
  const _Float16* r = BT + (n0 + (lane & 15)) * ldk + k0 + ((lane >> 4) << 4);
  v8h lo = *(const v8h*)r;
  v8h hi = *(const v8h*)(r + 8);
  return __builtin_shufflevector(lo, hi, 0, 1, 2, 3, 4, 5, 6, 7,
                                 8, 9, 10, 11, 12, 13, 14, 15);
}

// Same, with a branchless row clamp for ragged N (k_fc: N = 1000).
__device__ __forceinline__ v16h load_bt_frag_clamp(const _Float16* BT, int ldk,
                                                   int k0, int n0, int nmax) {
  const int lane = threadIdx.x & 31;
  int n = n0 + (lane & 15);
  n = (n > nmax) ? nmax : n;
  const _Float16* r = BT + n * ldk + k0 + ((lane >> 4) << 4);
  v8h lo = *(const v8h*)r;
  v8h hi = *(const v8h*)(r + 8);
  return __builtin_shufflevector(lo, hi, 0, 1, 2, 3, 4, 5, 6, 7,
                                 8, 9, 10, 11, 12, 13, 14, 15);
}

// ---------------------------------------------------------------------------
// Weight preparation kernels.
// k_cvtT:  fp32 [K x N] row-major  ->  f16 B^T [N x K] row-major.
// k_cvt_pad: fp32 [C x K] (already B^T-shaped) -> f16 [C x Kpad], zero pad.
// ---------------------------------------------------------------------------
__global__ void k_cvtT(const float* __restrict__ s, _Float16* __restrict__ d,
                       int K, int N) {
  const int total = K * N;
  for (int i = blockIdx.x * blockDim.x + threadIdx.x; i < total;
       i += gridDim.x * blockDim.x) {
    int k = i / N, n = i - k * N;
    d[n * K + k] = (_Float16)s[i];
  }
}

__global__ void k_cvt_pad(const float* __restrict__ s, _Float16* __restrict__ d,
                          int C, int K, int Kpad) {
  const int total = C * Kpad;
  for (int i = blockIdx.x * blockDim.x + threadIdx.x; i < total;
       i += gridDim.x * blockDim.x) {
    int c = i / Kpad, k = i - c * Kpad;
    d[i] = (k < K) ? (_Float16)s[c * K + k] : (_Float16)0.f;
  }
}

// ---------------------------------------------------------------------------
// Patch embedding + channel LayerNorm.
// Block = 256 thr (8 waves): 16 spatial positions x all C outputs.
// ---------------------------------------------------------------------------
__global__ __launch_bounds__(256) void k_patch_ln(
    const float* __restrict__ xin, const _Float16* __restrict__ pwT,
    const float* __restrict__ pnorm, float* __restrict__ out,
    int Cin, int Hin, int Win, int C, int hn, int wn, int K, int Kpad, int ps)
{
  __shared__ alignas(16) _Float16 As[16 * 640];
  __shared__ float Os[16 * 320];
  __shared__ float mv[16], vv[16];
  const int m0  = blockIdx.x * 16;
  const int tid = threadIdx.x;
  const int ps2 = ps * ps;

  for (int idx = tid; idx < 16 * Kpad; idx += 256) {
    int r = idx / Kpad, k = idx - r * Kpad;
    float v = 0.f;
    if (k < K) {
      int p   = m0 + r;
      int b   = p / (hn * wn);
      int rem = p - b * (hn * wn);
      int hc  = rem / wn, wc = rem - hc * wn;
      int ci  = k / ps2;
      int pr  = k - ci * ps2;
      int py  = pr / ps, px = pr - py * ps;
      v = xin[((b * Cin + ci) * Hin + hc * ps + py) * Win + wc * ps + px];
    }
    As[r * Kpad + k] = (_Float16)v;
  }
  __syncthreads();

  const int wave = tid >> 5, lane = tid & 31;
  const int ntiles = C >> 4;
  for (int nt = wave; nt < ntiles; nt += 8) {
    v8f acc = {};
    for (int k0 = 0; k0 < Kpad; k0 += 32) {
      v16h a = load_a_frag(As, Kpad, k0);
      v16h b = load_bt_frag(pwT, Kpad, k0, nt * 16);   // zero-padded weights
      acc = wmma16(a, b, acc);
    }
    int n  = nt * 16 + (lane & 15);
    int mo = (lane >> 4) * 8;
#pragma unroll
    for (int v = 0; v < 8; ++v) Os[(mo + v) * C + n] = acc[v];
  }
  __syncthreads();

  if (tid < 16) {
    float s1 = 0.f, s2 = 0.f;
    for (int c = 0; c < C; ++c) { float x = Os[tid * C + c]; s1 += x; s2 += x * x; }
    float m = s1 / C;
    mv[tid] = m;
    vv[tid] = rsqrtf(s2 / C - m * m + 1e-5f);
  }
  __syncthreads();

  for (int idx = tid; idx < 16 * C; idx += 256) {
    int r = idx / C, c = idx - r * C;
    int p = m0 + r;
    int b = p / (hn * wn);
    int rem = p - b * (hn * wn);
    int hc = rem / wn, wc = rem - hc * wn;
    out[((b * C + c) * hn + hc) * wn + wc] = (Os[idx] - mv[r]) * vv[r] * pnorm[c];
  }
}

// ---------------------------------------------------------------------------
// shift2d + window partition -> token-major f16 [nwin*64, C]
// ---------------------------------------------------------------------------
__global__ void k_shiftwin(const float* __restrict__ x, _Float16* __restrict__ xw,
                           int C, int H, int W, int hw, int ww)
{
  const int total = 16 * H * W * C;
  const int g = C / 12;
  for (int e = blockIdx.x * blockDim.x + threadIdx.x; e < total;
       e += gridDim.x * blockDim.x) {
    int tok = e / C, ch = e - tok * C;
    int wi = tok >> 6, t = tok & 63;
    int wy = t >> 3, wx = t & 7;
    int wc = wi % ww;
    int tmp = wi / ww;
    int hc = tmp % hw;
    int b  = tmp / hw;
    int y  = hc * 8 + wy, xx = wc * 8 + wx;
    int sy = y, sx = xx;
    if      (ch <     g) sx = xx + 1;   // shift (0,-1)
    else if (ch < 2 * g) sx = xx - 1;   // shift (0, 1)
    else if (ch < 3 * g) sy = y + 1;    // shift (-1,0)
    else if (ch < 4 * g) sy = y - 1;    // shift ( 1,0)
    float v = 0.f;
    if (sy >= 0 && sy < H && sx >= 0 && sx < W)
      v = x[((b * C + ch) * H + sy) * W + sx];
    xw[e] = (_Float16)v;
  }
}

// ---------------------------------------------------------------------------
// Fused projection: q/kv = xw @ w_fused[:, :2C].  q,kv scattered per head
// ([win][head][64][32]); kv additionally stored transposed ([win][head][32][64])
// so the attention P@KV GEMM can use contiguous B^T fragments.
// ---------------------------------------------------------------------------
__global__ __launch_bounds__(256) void k_proj_qkv(
    const _Float16* __restrict__ xw, const _Float16* __restrict__ wfT,
    _Float16* __restrict__ q, _Float16* __restrict__ kv,
    _Float16* __restrict__ kvT, int Mtok, int C, int heads)
{
  const int ntl  = (2 * C) >> 4;
  const int tile = blockIdx.x * 8 + (threadIdx.x >> 5);
  if (tile >= (Mtok >> 4) * ntl) return;
  const int tm = tile / ntl, tn = tile - tm * ntl;

  v8f acc = {};
  for (int k0 = 0; k0 < C; k0 += 32) {
    v16h a = load_a_frag(xw + tm * 16 * C, C, k0);
    v16h b = load_bt_frag(wfT, C, k0, tn * 16);
    acc = wmma16(a, b, acc);
  }
  const int lane = threadIdx.x & 31;
  int n  = tn * 16 + (lane & 15);
  int mo = tm * 16 + (lane >> 4) * 8;
  int isKV = (n >= C);
  int col  = isKV ? n - C : n;
  int head = col >> 5, d = col & 31;
#pragma unroll
  for (int v = 0; v < 8; ++v) {
    int gm = mo + v;
    int wi = gm >> 6, t = gm & 63;
    int base = (wi * heads + head) << 6;   // * 64
    _Float16 hv = (_Float16)acc[v];
    if (isKV) {
      kv [(base + t) * 32 + d] = hv;
      kvT[(base >> 1) * 64 + d * 64 + t] = hv;   // (wi*heads+head)*32*64
    } else {
      q  [(base + t) * 32 + d] = hv;
    }
  }
}

// ---------------------------------------------------------------------------
// Head gating: sigmoid(xw @ w_gate + b_gate)   (tiny: heads <= 10)
// ---------------------------------------------------------------------------
__global__ void k_gate(const _Float16* __restrict__ xw, const float* __restrict__ wg,
                       const float* __restrict__ bg, float* __restrict__ gate,
                       int Mtok, int C, int heads)
{
  const int total = Mtok * heads;
  for (int i = blockIdx.x * blockDim.x + threadIdx.x; i < total;
       i += gridDim.x * blockDim.x) {
    int tok = i / heads, h = i - tok * heads;
    float s = bg[h];
    const _Float16* r = xw + tok * C;
    for (int k = 0; k < C; ++k) s += (float)r[k] * wg[k * heads + h];
    gate[i] = 1.f / (1.f + expf(-s));
  }
}

// ---------------------------------------------------------------------------
// Per-(window, head) attention: S = Q KV^T * scale -> softmax -> xa = P KV,
// gated, scattered to xa16[tok][head*32+d].  Block = 256 thr / 8 waves.
// Both GEMMs read contiguous B^T fragments (kv rows / kvT rows).
// ---------------------------------------------------------------------------
__global__ __launch_bounds__(256) void k_attn(
    const _Float16* __restrict__ q, const _Float16* __restrict__ kv,
    const _Float16* __restrict__ kvT, const float* __restrict__ gate,
    _Float16* __restrict__ xa, int heads, int C)
{
  __shared__ float Ss[64 * 64];
  __shared__ alignas(16) _Float16 Ps[64 * 64];
  const int wi = blockIdx.x / heads, h = blockIdx.x - wi * heads;
  const int wh = wi * heads + h;
  const _Float16* Q  = q   + (wh << 6) * 32;
  const _Float16* KV = kv  + (wh << 6) * 32;
  const _Float16* KT = kvT + (wh << 5) * 64;
  const int wave = threadIdx.x >> 5, lane = threadIdx.x & 31;

  // ---- scores: 16 tiles of 16x16, K = 32 (one WMMA each) ----
  for (int tt = wave; tt < 16; tt += 8) {
    int tm = tt >> 2, tn = tt & 3;
    v16h a = load_a_frag(Q + tm * 16 * 32, 32, 0);
    v16h b = load_bt_frag(KV, 32, 0, tn * 16);       // B = KV^T (kv rows)
    v8f acc = {};
    acc = wmma16(a, b, acc);
    int n  = tn * 16 + (lane & 15);
    int mo = tm * 16 + (lane >> 4) * 8;
#pragma unroll
    for (int v = 0; v < 8; ++v) Ss[(mo + v) * 64 + n] = acc[v];
  }
  __syncthreads();

  // ---- softmax over rows (scale = DH^-0.5) ----
  if (threadIdx.x < 64) {
    const int r = threadIdx.x;
    const float scl = 0.17677669529663689f;
    float mx = -1e30f;
    for (int j = 0; j < 64; ++j) mx = fmaxf(mx, Ss[r * 64 + j] * scl);
    float sum = 0.f;
    for (int j = 0; j < 64; ++j) {
      float e = expf(Ss[r * 64 + j] * scl - mx);
      Ss[r * 64 + j] = e;
      sum += e;
    }
    float inv = 1.f / sum;
    for (int j = 0; j < 64; ++j) Ps[r * 64 + j] = (_Float16)(Ss[r * 64 + j] * inv);
  }
  __syncthreads();

  // ---- xa = P @ KV : 8 tiles (4 M x 2 N), one per wave ----
  {
    int tm = wave >> 1, tn = wave & 1;
    v8f acc = {};
    for (int k0 = 0; k0 < 64; k0 += 32) {
      v16h a = load_a_frag(Ps + tm * 16 * 64, 64, k0);
      v16h b = load_bt_frag(KT, 64, k0, tn * 16);    // B^T = kvT rows (d-major)
      acc = wmma16(a, b, acc);
    }
    int d  = tn * 16 + (lane & 15);
    int mo = tm * 16 + (lane >> 4) * 8;
#pragma unroll
    for (int v = 0; v < 8; ++v) {
      int gtok = (wi << 6) + mo + v;
      float g = gate[gtok * heads + h];
      xa[gtok * C + h * 32 + d] = (_Float16)(acc[v] * g);
    }
  }
}

// ---------------------------------------------------------------------------
// y = xa @ w_out + star_relu(xw @ w_fused[:,2C:6C]) @ w_ff, then channel
// LayerNorm (norm_w) and residual add; un-windowed write to NCHW fp32.
// Block = 16 tokens; star-relu'd FF activations staged in LDS.
// ---------------------------------------------------------------------------
__global__ __launch_bounds__(256) void k_ffout(
    const _Float16* __restrict__ xw, const _Float16* __restrict__ xa,
    const _Float16* __restrict__ wfT, const _Float16* __restrict__ woT,
    const _Float16* __restrict__ wffT, const float* __restrict__ srs,
    const float* __restrict__ srb, const float* __restrict__ nw,
    const float* __restrict__ resid, float* __restrict__ out,
    int C, int hw, int ww, int H, int W)
{
  __shared__ alignas(16) _Float16 FFs[16 * 1280];
  __shared__ float Os[16 * 320];
  __shared__ float mv[16], vv[16];
  const int m0   = blockIdx.x * 16;
  const int wave = threadIdx.x >> 5, lane = threadIdx.x & 31;
  const int dff  = 4 * C;
  const float s  = srs[0], bb = srb[0];

  // phase 1: FF = star_relu(xw @ Wf[:, 2C:6C])  -> LDS (f16)
  for (int nt = wave; nt < (dff >> 4); nt += 8) {
    v8f acc = {};
    for (int k0 = 0; k0 < C; k0 += 32) {
      v16h a = load_a_frag(xw + m0 * C, C, k0);
      v16h b = load_bt_frag(wfT, C, k0, 2 * C + nt * 16);
      acc = wmma16(a, b, acc);
    }
    int n  = nt * 16 + (lane & 15);
    int mo = (lane >> 4) * 8;
#pragma unroll
    for (int v = 0; v < 8; ++v) {
      float r = fmaxf(acc[v], 0.f);
      FFs[(mo + v) * dff + n] = (_Float16)(s * r * r + bb);
    }
  }
  __syncthreads();

  // phase 2: y = xa @ w_out + FF @ w_ff
  for (int nt = wave; nt < (C >> 4); nt += 8) {
    v8f acc = {};
    for (int k0 = 0; k0 < C; k0 += 32) {
      v16h a = load_a_frag(xa + m0 * C, C, k0);
      v16h b = load_bt_frag(woT, C, k0, nt * 16);
      acc = wmma16(a, b, acc);
    }
    for (int k0 = 0; k0 < dff; k0 += 32) {
      v16h a = load_a_frag(FFs, dff, k0);
      v16h b = load_bt_frag(wffT, dff, k0, nt * 16);
      acc = wmma16(a, b, acc);
    }
    int n  = nt * 16 + (lane & 15);
    int mo = (lane >> 4) * 8;
#pragma unroll
    for (int v = 0; v < 8; ++v) Os[(mo + v) * C + n] = acc[v];
  }
  __syncthreads();

  if (threadIdx.x < 16) {
    float s1 = 0.f, s2 = 0.f;
    for (int c = 0; c < C; ++c) { float x = Os[threadIdx.x * C + c]; s1 += x; s2 += x * x; }
    float m = s1 / C;
    mv[threadIdx.x] = m;
    vv[threadIdx.x] = rsqrtf(s2 / C - m * m + 1e-5f);
  }
  __syncthreads();

  for (int idx = threadIdx.x; idx < 16 * C; idx += 256) {
    int r = idx / C, c = idx - r * C;
    int tok = m0 + r;
    int wi = tok >> 6, t = tok & 63;
    int wy = t >> 3, wx = t & 7;
    int wc = wi % ww;
    int tmp = wi / ww;
    int hc = tmp % hw;
    int b  = tmp / hw;
    int gi = ((b * C + c) * H + hc * 8 + wy) * W + wc * 8 + wx;
    out[gi] = resid[gi] + (Os[idx] - mv[r]) * vv[r] * nw[c];
  }
}

// ---------------------------------------------------------------------------
// PEG: depthwise 3x3 conv (pad 1) residual
// ---------------------------------------------------------------------------
__global__ void k_peg(const float* __restrict__ x, const float* __restrict__ kk,
                      float* __restrict__ y, int C, int H, int W)
{
  const int total = 16 * C * H * W;
  for (int i = blockIdx.x * blockDim.x + threadIdx.x; i < total;
       i += gridDim.x * blockDim.x) {
    int xw_ = i % W;
    int t = i / W;
    int yh = t % H; t /= H;
    int c  = t % C;
    int b  = t / C;
    const float* kr = kk + c * 9;
    float acc = 0.f;
#pragma unroll
    for (int dy = 0; dy < 3; ++dy)
#pragma unroll
      for (int dx = 0; dx < 3; ++dx) {
        int sy = yh + dy - 1, sx = xw_ + dx - 1;
        if (sy >= 0 && sy < H && sx >= 0 && sx < W)
          acc += kr[dy * 3 + dx] * x[((b * C + c) * H + sy) * W + sx];
      }
    y[i] = x[i] + acc;
  }
}

// ---------------------------------------------------------------------------
// Head: global average pool, LayerNorm, classifier GEMM (M=16, K=320, N=1000)
// ---------------------------------------------------------------------------
__global__ void k_pool(const float* __restrict__ x, float* __restrict__ p,
                       int C, int HW)
{
  int i = blockIdx.x * blockDim.x + threadIdx.x;
  if (i >= 16 * C) return;
  const float* r = x + i * HW;
  float s = 0.f;
  for (int j = 0; j < HW; ++j) s += r[j];
  p[i] = s / HW;
}

__global__ __launch_bounds__(256) void k_headln(const float* __restrict__ p,
                                                const float* __restrict__ hnw,
                                                _Float16* __restrict__ xn, int C)
{
  __shared__ float rs[256], rs2[256];
  const int b = blockIdx.x;
  float s = 0.f, s2 = 0.f;
  for (int c = threadIdx.x; c < C; c += 256) {
    float v = p[b * C + c];
    s += v; s2 += v * v;
  }
  rs[threadIdx.x] = s; rs2[threadIdx.x] = s2;
  __syncthreads();
  for (int o = 128; o > 0; o >>= 1) {
    if (threadIdx.x < o) {
      rs[threadIdx.x]  += rs[threadIdx.x + o];
      rs2[threadIdx.x] += rs2[threadIdx.x + o];
    }
    __syncthreads();
  }
  float m = rs[0] / C;
  float inv = rsqrtf(rs2[0] / C - m * m + 1e-5f);
  for (int c = threadIdx.x; c < C; c += 256)
    xn[b * C + c] = (_Float16)((p[b * C + c] - m) * inv * hnw[c]);
}

__global__ __launch_bounds__(256) void k_fc(const _Float16* __restrict__ xn,
                                            const _Float16* __restrict__ wT,
                                            float* __restrict__ out, int C, int N)
{
  const int ntl  = (N + 15) >> 4;
  const int tile = blockIdx.x * 8 + (threadIdx.x >> 5);
  if (tile >= ntl) return;
  const int lane = threadIdx.x & 31;
  v8f acc = {};
  for (int k0 = 0; k0 < C; k0 += 32) {
    v16h a = load_a_frag(xn, C, k0);
    v16h b = load_bt_frag_clamp(wT, C, k0, tile * 16, N - 1);
    acc = wmma16(a, b, acc);
  }
  int n  = tile * 16 + (lane & 15);
  int mo = (lane >> 4) * 8;
  if (n < N) {
#pragma unroll
    for (int v = 0; v < 8; ++v) out[(mo + v) * N + n] = acc[v];
  }
}

// ---------------------------------------------------------------------------
// Host driver.  Input flattening assumption: recursive insertion order of the
// setup_inputs() pytree:
//   x, then per stage: pe_w, pe_norm, peg, per block {w_fused, w_gate, b_gate,
//   w_out, sr_s, sr_b, w_ff, norm_w}; then head_norm, fc_w.
// ---------------------------------------------------------------------------
extern "C" void kernel_launch(void* const* d_in, const int* in_sizes, int n_in,
                              void* d_out, int out_size, void* d_ws, size_t ws_size,
                              hipStream_t stream)
{
  (void)in_sizes; (void)n_in; (void)out_size; (void)ws_size;
  static const int DEP[4] = {2, 2, 6, 2};
  static const int DIM[4] = {64, 128, 160, 320};
  static const int PAT[4] = {4, 2, 2, 2};

  int idx = 0;
  const float* x_in = (const float*)d_in[idx++];
  const float *pe_w[4], *pe_norm[4], *pegw[4];
  const float *wfu[4][6], *wga[4][6], *bga[4][6], *wou[4][6];
  const float *srs[4][6], *srb[4][6], *wff[4][6], *nrw[4][6];
  for (int si = 0; si < 4; ++si) {
    pe_w[si]    = (const float*)d_in[idx++];
    pe_norm[si] = (const float*)d_in[idx++];
    pegw[si]    = (const float*)d_in[idx++];
    for (int bi = 0; bi < DEP[si]; ++bi) {
      wfu[si][bi] = (const float*)d_in[idx++];
      wga[si][bi] = (const float*)d_in[idx++];
      bga[si][bi] = (const float*)d_in[idx++];
      wou[si][bi] = (const float*)d_in[idx++];
      srs[si][bi] = (const float*)d_in[idx++];
      srb[si][bi] = (const float*)d_in[idx++];
      wff[si][bi] = (const float*)d_in[idx++];
      nrw[si][bi] = (const float*)d_in[idx++];
    }
  }
  const float* head_norm = (const float*)d_in[idx++];
  const float* fc_w      = (const float*)d_in[idx++];

  // ---- workspace carve-up (256B aligned) ----
  size_t off = 0;
  char* base = (char*)d_ws;
  auto carve = [&](size_t bytes) -> void* {
    void* p = base + off;
    off = (off + bytes + 255) & ~(size_t)255;
    return p;
  };
  const size_t MAXACT = (size_t)16 * 64 * 64 * 64;       // 4.19M elems
  float*    bufA   = (float*)carve(MAXACT * 4);
  float*    bufB   = (float*)carve(MAXACT * 4);
  _Float16* xw16   = (_Float16*)carve(MAXACT * 2);
  _Float16* q16    = (_Float16*)carve(MAXACT * 2);
  _Float16* kv16   = (_Float16*)carve(MAXACT * 2);
  _Float16* kvT16  = (_Float16*)carve(MAXACT * 2);
  _Float16* xa16   = (_Float16*)carve(MAXACT * 2);
  float*    gateB  = (float*)carve((size_t)131072 * 4);
  _Float16* wfT16  = (_Float16*)carve((size_t)320 * 1920 * 2);
  _Float16* woT16  = (_Float16*)carve((size_t)320 * 320 * 2);
  _Float16* wffT16 = (_Float16*)carve((size_t)1280 * 320 * 2);
  _Float16* peT16  = (_Float16*)carve((size_t)320 * 640 * 2);
  _Float16* fcT16  = (_Float16*)carve((size_t)1000 * 320 * 2);
  _Float16* xn16   = (_Float16*)carve((size_t)16 * 320 * 2);
  float*    poolB  = (float*)carve((size_t)16 * 320 * 4);

  auto cvtT = [&](const float* s, _Float16* d, int K, int N) {
    k_cvtT<<<(K * N + 255) / 256, 256, 0, stream>>>(s, d, K, N);
  };

  int H = 256, W = 256, Cin = 3;
  float* cur = bufA;
  float* oth = bufB;
  const float* stageIn = x_in;

  for (int si = 0; si < 4; ++si) {
    const int ps = PAT[si], C = DIM[si];
    const int hn = H / ps, wn = W / ps;
    const int K = Cin * ps * ps;
    const int Kpad = (K + 31) & ~31;

    k_cvt_pad<<<(C * Kpad + 255) / 256, 256, 0, stream>>>(pe_w[si], peT16,
                                                          C, K, Kpad);
    const int M = 16 * hn * wn;
    k_patch_ln<<<M / 16, 256, 0, stream>>>(stageIn, peT16, pe_norm[si], cur,
                                           Cin, H, W, C, hn, wn, K, Kpad, ps);
    H = hn; W = wn;
    const int hw = H / 8, ww = W / 8;
    const int Mtok = 16 * H * W;
    const int heads = C / 32;
    const int dff = 4 * C;
    const int nwin = Mtok / 64;

    for (int bi = 0; bi < DEP[si]; ++bi) {
      if (bi == 1) {
        int tot = 16 * C * H * W;
        k_peg<<<(tot + 255) / 256, 256, 0, stream>>>(cur, pegw[si], oth, C, H, W);
        float* t = cur; cur = oth; oth = t;
      }
      cvtT(wfu[si][bi], wfT16, C, 6 * C);     // -> [6C x C]
      cvtT(wou[si][bi], woT16, C, C);         // -> [C x C]
      cvtT(wff[si][bi], wffT16, dff, C);      // -> [C x dff]

      {
        int tot = Mtok * C;
        k_shiftwin<<<(tot + 255) / 256, 256, 0, stream>>>(cur, xw16, C, H, W, hw, ww);
      }
      {
        int tiles = (Mtok / 16) * ((2 * C) / 16);
        k_proj_qkv<<<(tiles + 7) / 8, 256, 0, stream>>>(xw16, wfT16, q16, kv16,
                                                        kvT16, Mtok, C, heads);
      }
      {
        int tot = Mtok * heads;
        k_gate<<<(tot + 255) / 256, 256, 0, stream>>>(xw16, wga[si][bi],
                                                      bga[si][bi], gateB,
                                                      Mtok, C, heads);
      }
      k_attn<<<nwin * heads, 256, 0, stream>>>(q16, kv16, kvT16, gateB, xa16,
                                               heads, C);
      k_ffout<<<Mtok / 16, 256, 0, stream>>>(xw16, xa16, wfT16, woT16, wffT16,
                                             srs[si][bi], srb[si][bi],
                                             nrw[si][bi], cur, oth,
                                             C, hw, ww, H, W);
      float* t = cur; cur = oth; oth = t;
    }
    Cin = C;
    stageIn = cur;
    if (si < 3) { float* t = cur; cur = oth; oth = t; }
  }

  // ---- head ----
  {
    const int C = 320, HW = H * W;   // H = W = 8 here
    k_pool<<<(16 * C + 255) / 256, 256, 0, stream>>>(stageIn, poolB, C, HW);
    k_headln<<<16, 256, 0, stream>>>(poolB, head_norm, xn16, C);
    cvtT(fc_w, fcT16, C, 1000);              // -> [1000 x 320]
    const int ntl = (1000 + 15) / 16;
    k_fc<<<(ntl + 7) / 8, 256, 0, stream>>>(xn16, fcT16, (float*)d_out, C, 1000);
  }
}
